// SpaceByteModel_89773406421098
// MI455X (gfx1250) — compile-verified, matrix operand order checked by reading, not run on hardware
//
#include <hip/hip_runtime.h>
#include <hip/hip_bf16.h>
#include <math.h>

// ---------------------------------------------------------------------------
// SpaceByte forward for MI455X (gfx1250, wave32, WMMA).
// All dense GEMMs run through v_wmma_f32_16x16x32_f16 (f16 A/B, f32 acc).
// Weights are read as fp32 and converted to f16 while staging into LDS
// (192MB L2 makes re-reading fp32 weights cheap; activations stay f16).
// Attention (tiny FLOP share) is a flash-style VALU kernel.
// ---------------------------------------------------------------------------

typedef __attribute__((ext_vector_type(16))) _Float16 v16h;
typedef __attribute__((ext_vector_type(8)))  _Float16 v8h;
typedef __attribute__((ext_vector_type(4)))  _Float16 v4h;
typedef __attribute__((ext_vector_type(8)))  float    v8f;
typedef __attribute__((ext_vector_type(4)))  float    v4f;

#define CFG_DL   512
#define CFG_DG   1024
#define CFG_NHL  8
#define CFG_NHG  16
#define CFG_NLL  4
#define CFG_NLG  8
#define CFG_P    32
#define CFG_B    4
#define CFG_T    4096
#define CFG_V    256
#define CFG_NP   (CFG_T / CFG_P)          // 128 patches per sequence
#define CFG_MG   (CFG_B * CFG_NP)         // 512 global tokens
#define CFG_ML   (CFG_B * CFG_T)          // 16384 local tokens
#define HD       64                       // head dim (both transformers)

// ---------------------------------------------------------------------------
// WMMA GEMM:  C[M,N] = A[M,K](f16, row-major) * W[N,K](f32, row-major)^T
// Block: 256 threads = 8 waves arranged 2(M) x 4(N); block tile 64x128.
// Each wave owns a 32x32 output strip: 2 A-frags x 2 B-frags -> 4 WMMA/K-step
// (4 matrix ops per 8 ds_load_b128 fragment loads).
// Fragment layout per ISA 7.12.2: per lane, the 16-half fragment is two
// contiguous 8-half (16B) chunks at k = {g*8..g*8+7} and {16+g*8..16+g*8+7}.
// ---------------------------------------------------------------------------
__device__ __forceinline__ v16h frag_from_lds(const _Float16* p) {
  v8h lo = *(const v8h*)p;
  v8h hi = *(const v8h*)(p + 16);
  v16h r;
#pragma unroll
  for (int i = 0; i < 8; ++i) { r[i] = lo[i]; r[i + 8] = hi[i]; }
  return r;
}

// mode: 0 = f32 out, 1 = f32 out + residual R, 2 = exact-GELU -> f16, 3 = f16
__device__ __forceinline__ void epi_store(int mode, void* C, const float* R,
                                          size_t i, float v) {
  switch (mode) {
    case 0: ((float*)C)[i] = v; break;
    case 1: ((float*)C)[i] = v + R[i]; break;
    case 2: ((_Float16*)C)[i] =
              (_Float16)(0.5f * v * (1.0f + erff(v * 0.70710678118f)));
            break;
    default: ((_Float16*)C)[i] = (_Float16)v; break;
  }
}

__global__ __launch_bounds__(256) void wmma_gemm(
    const _Float16* __restrict__ A, const float* __restrict__ Bw,
    void* C, const float* R, int M, int N, int K, int mode) {
  constexpr int LDT = 40;                    // padded LDS row stride (halves)
  __shared__ _Float16 As[64 * LDT];          // 64 x 32 f16 tile of A
  __shared__ _Float16 Bs[128 * LDT];         // 128 x 32 f16 tile of W

  const int tid  = threadIdx.x;
  const int lane = tid & 31;
  const int wave = tid >> 5;
  const int wr   = wave >> 2;                // 0..1 (wave row, 32 M each)
  const int wc   = wave & 3;                 // 0..3 (wave col, 32 N each)
  const int m0   = blockIdx.y * 64;
  const int n0   = blockIdx.x * 128;
  const int fr   = lane & 15;
  const int fg   = lane >> 4;

  // A-stage assignment: 64 rows x 4 chunks(16B) = 256 -> one per thread.
  const int ar = tid >> 2, ac = tid & 3;
  const _Float16* a_src = A + (size_t)(m0 + ar) * K + ac * 8;
  _Float16* a_dst = As + ar * LDT + ac * 8;

  // B-stage assignment: 128 rows x 8 chunks(4 floats) = 1024 -> 4 per thread.
  const int br = tid >> 3, bc = tid & 7;     // + q*32 rows per sub-step
  const float* b_src[4];
  _Float16* b_dst[4];
#pragma unroll
  for (int q = 0; q < 4; ++q) {
    const int r = br + q * 32;
    b_src[q] = Bw + (size_t)(n0 + r) * K + bc * 4;
    b_dst[q] = Bs + r * LDT + bc * 4;
  }

  v8f acc00 = {}, acc01 = {}, acc10 = {}, acc11 = {};

  for (int k0 = 0; k0 < K; k0 += 32) {
    // Phase 1: issue all global loads (1x A f16 chunk + 4x W f32 chunks) so
    // they overlap; a single s_wait covers them before the stores.
    const v8h a_val = *(const v8h*)(a_src + k0);
    v4f f[4];
#pragma unroll
    for (int q = 0; q < 4; ++q) f[q] = *(const v4f*)(b_src[q] + k0);

    // One cacheline prefetch per next-step row slice (128B contiguous).
    if (k0 + 32 < K && bc == 0) {
#pragma unroll
      for (int q = 0; q < 4; ++q) __builtin_prefetch(b_src[q] + k0 + 32, 0, 1);
    }

    // Phase 2: convert + store to LDS.
    *(v8h*)a_dst = a_val;
#pragma unroll
    for (int q = 0; q < 4; ++q) {
      v4h h;
      h[0] = (_Float16)f[q][0]; h[1] = (_Float16)f[q][1];
      h[2] = (_Float16)f[q][2]; h[3] = (_Float16)f[q][3];
      *(v4h*)b_dst[q] = h;
    }
    __syncthreads();

    v16h a0 = frag_from_lds(As + (wr * 32 + fr) * LDT + fg * 8);
    v16h a1 = frag_from_lds(As + (wr * 32 + 16 + fr) * LDT + fg * 8);
    v16h b0 = frag_from_lds(Bs + (wc * 32 + fr) * LDT + fg * 8);
    v16h b1 = frag_from_lds(Bs + (wc * 32 + 16 + fr) * LDT + fg * 8);
    acc00 = __builtin_amdgcn_wmma_f32_16x16x32_f16(false, a0, false, b0,
                                                   (short)0, acc00, false, false);
    acc01 = __builtin_amdgcn_wmma_f32_16x16x32_f16(false, a0, false, b1,
                                                   (short)0, acc01, false, false);
    acc10 = __builtin_amdgcn_wmma_f32_16x16x32_f16(false, a1, false, b0,
                                                   (short)0, acc10, false, false);
    acc11 = __builtin_amdgcn_wmma_f32_16x16x32_f16(false, a1, false, b1,
                                                   (short)0, acc11, false, false);
    __syncthreads();
  }

  // C layout (ISA 7.12.2): vgpr r -> M = r + (lane>=16 ? 8 : 0), N = lane&15.
  const int mbase0 = m0 + wr * 32 + fg * 8;
  const int mbase1 = mbase0 + 16;
  const int na = n0 + wc * 32 + fr;
#pragma unroll
  for (int r = 0; r < 8; ++r) {
    const size_t row0 = (size_t)(mbase0 + r) * N;
    const size_t row1 = (size_t)(mbase1 + r) * N;
    epi_store(mode, C, R, row0 + na, acc00[r]);
    epi_store(mode, C, R, row0 + na + 16, acc01[r]);
    epi_store(mode, C, R, row1 + na, acc10[r]);
    epi_store(mode, C, R, row1 + na + 16, acc11[r]);
  }
}

// ---------------------------------------------------------------------------
// Flash attention: one block per (batch, head), one query row per thread.
// qkv rows are [q | k | v] each of width d; head h uses columns h*64..h*64+63.
// K/V staged in LDS; online-softmax accumulation in registers (hd = 64).
// ---------------------------------------------------------------------------
__global__ __launch_bounds__(128) void attn_flash(
    const _Float16* __restrict__ qkv, _Float16* __restrict__ o,
    int T, int nh, int d) {
  __shared__ _Float16 Ksh[128 * HD];
  __shared__ _Float16 Vsh[128 * HD];
  const int head = blockIdx.x;
  const int b = head / nh, h = head % nh;
  const int t = threadIdx.x;                 // blockDim.x == T
  const size_t rb = (size_t)(b * T + t) * (3 * d);

  const _Float16* kp = qkv + rb + d + h * HD;
  const _Float16* vp = kp + d;
#pragma unroll
  for (int c = 0; c < HD; c += 8) {
    *(v8h*)(Ksh + t * HD + c) = *(const v8h*)(kp + c);
    *(v8h*)(Vsh + t * HD + c) = *(const v8h*)(vp + c);
  }
  __syncthreads();

  float q[HD];
  const _Float16* qp = qkv + rb + h * HD;
#pragma unroll
  for (int c = 0; c < HD; ++c) q[c] = (float)qp[c];

  float mrun = -3.0e38f, l = 0.0f, acc[HD];
#pragma unroll
  for (int c = 0; c < HD; ++c) acc[c] = 0.0f;

  for (int j = 0; j <= t; ++j) {             // causal
    float s = 0.0f;
#pragma unroll
    for (int c = 0; c < HD; ++c) s += q[c] * (float)Ksh[j * HD + c];
    s *= 0.125f;                             // 1/sqrt(64)
    const float mn = fmaxf(mrun, s);
    const float corr = __expf(mrun - mn);
    const float p = __expf(s - mn);
    l = l * corr + p;
#pragma unroll
    for (int c = 0; c < HD; ++c)
      acc[c] = acc[c] * corr + p * (float)Vsh[j * HD + c];
    mrun = mn;
  }
  _Float16* op = o + (size_t)(b * T + t) * d + h * HD;
  const float inv = 1.0f / l;
#pragma unroll
  for (int c = 0; c < HD; ++c) op[c] = (_Float16)(acc[c] * inv);
}

// ---------------------------------------------------------------------------
// LayerNorm: f32 in -> f16 out (feeds next WMMA GEMM). One block per row.
// ---------------------------------------------------------------------------
__global__ __launch_bounds__(256) void layernorm_f16(
    const float* __restrict__ x, const float* __restrict__ g,
    const float* __restrict__ bb, _Float16* __restrict__ y, int d) {
  __shared__ float red[256];
  const int row = blockIdx.x, tid = threadIdx.x;
  const float* xr = x + (size_t)row * d;

  float s = 0.0f;
  for (int i = tid; i < d; i += 256) s += xr[i];
  red[tid] = s; __syncthreads();
  for (int off = 128; off > 0; off >>= 1) {
    if (tid < off) red[tid] += red[tid + off];
    __syncthreads();
  }
  const float mu = red[0] / d; __syncthreads();

  float v = 0.0f;
  for (int i = tid; i < d; i += 256) { float t = xr[i] - mu; v += t * t; }
  red[tid] = v; __syncthreads();
  for (int off = 128; off > 0; off >>= 1) {
    if (tid < off) red[tid] += red[tid + off];
    __syncthreads();
  }
  const float rstd = rsqrtf(red[0] / d + 1e-5f);

  _Float16* yr = y + (size_t)row * d;
  for (int i = tid; i < d; i += 256)
    yr[i] = (_Float16)((xr[i] - mu) * rstd * g[i] + bb[i]);
}

// ---------------------------------------------------------------------------
// Embedding gather + per-patch mean. One block per patch (32 bytes x 512 dims).
// Writes e (f16, kept for local_in assembly) and patch means (f16 GEMM input).
// ---------------------------------------------------------------------------
__global__ __launch_bounds__(256) void embed_pool(
    const int* __restrict__ ids, const float* __restrict__ emb,
    _Float16* __restrict__ e, _Float16* __restrict__ pv) {
  const int g = blockIdx.x;                  // global patch index, b*128 + j
  const int tid = threadIdx.x;
  __shared__ int sid[CFG_P];
  if (tid < CFG_P) sid[tid] = ids[g * CFG_P + tid];
  __syncthreads();
  for (int dcol = tid; dcol < CFG_DL; dcol += 256) {
    float s = 0.0f;
#pragma unroll 4
    for (int p = 0; p < CFG_P; ++p) {
      const float v = emb[(size_t)sid[p] * CFG_DL + dcol];
      e[((size_t)g * CFG_P + p) * CFG_DL + dcol] = (_Float16)v;
      s += v;
    }
    pv[(size_t)g * CFG_DL + dcol] = (_Float16)(s * (1.0f / CFG_P));
  }
}

// local_in[g] = [ g2l[g] ; bytes 0..30 of patch g ]  (f32 local-transformer x)
__global__ __launch_bounds__(256) void build_local_in(
    const float* __restrict__ g2l, const _Float16* __restrict__ e,
    float* __restrict__ lx) {
  const int g = blockIdx.x;
  for (int i = threadIdx.x; i < CFG_P * CFG_DL; i += 256) {
    const int pos = i >> 9, dcol = i & (CFG_DL - 1);
    float v;
    if (pos == 0) v = g2l[(size_t)g * CFG_DL + dcol];
    else v = (float)e[((size_t)g * CFG_P + (pos - 1)) * CFG_DL + dcol];
    lx[((size_t)g * CFG_P + pos) * CFG_DL + dcol] = v;
  }
}

// ---------------------------------------------------------------------------
// Host-side orchestration.
// ---------------------------------------------------------------------------
struct LayerP {
  const float *Wff1, *Wff2, *Wout, *Wqkv, *ln1_b, *ln1_g, *ln2_b, *ln2_g;
};

static char* arena_take(void* ws, size_t& off, size_t bytes) {
  char* p = (char*)ws + off;
  off += (bytes + 255) & ~(size_t)255;
  return p;
}

extern "C" void kernel_launch(void* const* d_in, const int* in_sizes, int n_in,
                              void* d_out, int out_size, void* d_ws,
                              size_t ws_size, hipStream_t stream) {
  (void)in_sizes; (void)out_size; (void)ws_size;
  if (n_in < 105) return;

  // jax tree_flatten order (dict keys sorted):
  //   byte_ids, W_g2l, W_out, W_patch, embed,
  //   global layers x8 {Wff1,Wff2,Wout,Wqkv,ln1_b,ln1_g,ln2_b,ln2_g}, lnf_b, lnf_g,
  //   local  layers x4 {...same...}, lnf_b, lnf_g
  const int*   byte_ids = (const int*)d_in[0];
  const float* W_g2l    = (const float*)d_in[1];
  const float* W_outp   = (const float*)d_in[2];
  const float* W_patch  = (const float*)d_in[3];
  const float* embed    = (const float*)d_in[4];

  LayerP G[CFG_NLG], L[CFG_NLL];
  int ix = 5;
  for (int i = 0; i < CFG_NLG; ++i, ix += 8) {
    G[i] = { (const float*)d_in[ix + 0], (const float*)d_in[ix + 1],
             (const float*)d_in[ix + 2], (const float*)d_in[ix + 3],
             (const float*)d_in[ix + 4], (const float*)d_in[ix + 5],
             (const float*)d_in[ix + 6], (const float*)d_in[ix + 7] };
  }
  const float* g_lnf_b = (const float*)d_in[ix++];
  const float* g_lnf_g = (const float*)d_in[ix++];
  for (int i = 0; i < CFG_NLL; ++i, ix += 8) {
    L[i] = { (const float*)d_in[ix + 0], (const float*)d_in[ix + 1],
             (const float*)d_in[ix + 2], (const float*)d_in[ix + 3],
             (const float*)d_in[ix + 4], (const float*)d_in[ix + 5],
             (const float*)d_in[ix + 6], (const float*)d_in[ix + 7] };
  }
  const float* l_lnf_b = (const float*)d_in[ix++];
  const float* l_lnf_g = (const float*)d_in[ix++];

  // Workspace arena (~205 MB). Every region is fully written before read.
  size_t off = 0;
  _Float16* e_f16 = (_Float16*)arena_take(d_ws, off, (size_t)CFG_ML * CFG_DL * 2);
  _Float16* pv    = (_Float16*)arena_take(d_ws, off, (size_t)CFG_MG * CFG_DL * 2);
  float*    xg    = (float*)   arena_take(d_ws, off, (size_t)CFG_MG * CFG_DG * 4);
  float*    g2l   = (float*)   arena_take(d_ws, off, (size_t)CFG_MG * CFG_DL * 4);
  float*    lx    = (float*)   arena_take(d_ws, off, (size_t)CFG_ML * CFG_DL * 4);
  // shared between phases (sized for the larger of global/local use):
  const size_t XN_ELTS  = (size_t)CFG_ML * CFG_DL;
  const size_t QKV_ELTS = (size_t)CFG_ML * 3 * CFG_DL;
  const size_t T_ELTS   = (size_t)CFG_ML * 4 * CFG_DL / 2 * 2;
  _Float16* xn  = (_Float16*)arena_take(d_ws, off, XN_ELTS * 2);
  _Float16* qkv = (_Float16*)arena_take(d_ws, off, QKV_ELTS * 2);
  _Float16* oat = (_Float16*)arena_take(d_ws, off, XN_ELTS * 2);
  _Float16* ff  = (_Float16*)arena_take(d_ws, off, T_ELTS * 2);

  auto gemm = [&](const _Float16* A, const float* Bw, void* C, const float* R,
                  int M, int N, int K, int mode) {
    dim3 grid(N / 128, M / 64);
    wmma_gemm<<<grid, 256, 0, stream>>>(A, Bw, C, R, M, N, K, mode);
  };

  auto tblock = [&](float* x, int M, int d, int nh, int T, const LayerP& p) {
    layernorm_f16<<<M, 256, 0, stream>>>(x, p.ln1_g, p.ln1_b, xn, d);
    gemm(xn, p.Wqkv, qkv, nullptr, M, 3 * d, d, 3);            // -> f16 qkv
    attn_flash<<<(M / T) * nh, T, 0, stream>>>(qkv, oat, T, nh, d);
    gemm(oat, p.Wout, x, x, M, d, d, 1);                       // + residual
    layernorm_f16<<<M, 256, 0, stream>>>(x, p.ln2_g, p.ln2_b, xn, d);
    gemm(xn, p.Wff1, ff, nullptr, M, 4 * d, d, 2);             // fused GELU
    gemm(ff, p.Wff2, x, x, M, d, 4 * d, 1);                    // + residual
  };

  // 1) Embedding gather + patch mean pooling.
  embed_pool<<<CFG_MG, 256, 0, stream>>>(byte_ids, embed, e_f16, pv);

  // 2) Patch projection into global width: x_g = pv @ W_patch^T.
  gemm(pv, W_patch, xg, nullptr, CFG_MG, CFG_DG, CFG_DL, 0);

  // 3) Global transformer (8 layers, d=1024, nh=16, T=128 patches).
  for (int i = 0; i < CFG_NLG; ++i)
    tblock(xg, CFG_MG, CFG_DG, CFG_NHG, CFG_NP, G[i]);

  // 4) Final global LN, project back to local width: g2l = h_g @ W_g2l^T.
  layernorm_f16<<<CFG_MG, 256, 0, stream>>>(xg, g_lnf_g, g_lnf_b, xn, CFG_DG);
  gemm(xn, W_g2l, g2l, nullptr, CFG_MG, CFG_DL, CFG_DG, 0);

  // 5) Assemble local inputs: [global ctx ; first 31 byte embeddings] / patch.
  build_local_in<<<CFG_MG, 256, 0, stream>>>(g2l, e_f16, lx);

  // 6) Local transformer (4 layers, d=512, nh=8, T=32), batch = 512 patches.
  for (int i = 0; i < CFG_NLL; ++i)
    tblock(lx, CFG_ML, CFG_DL, CFG_NHL, CFG_P, L[i]);

  // 7) Final local LN + vocab projection straight into d_out (f32 logits).
  layernorm_f16<<<CFG_ML, 256, 0, stream>>>(lx, l_lnf_g, l_lnf_b, xn, CFG_DL);
  gemm(xn, W_outp, d_out, nullptr, CFG_ML, CFG_V, CFG_DL, 0);
}
